// X_LSTM_50766513438976
// MI455X (gfx1250) — matrix operand order repaired and checked by reading, not compiled, not run on hardware
//
#include <hip/hip_runtime.h>

typedef __bf16 bf16;
typedef __attribute__((ext_vector_type(16))) __bf16 bf16x16;
typedef __attribute__((ext_vector_type(8)))  float  f32x8;
typedef unsigned int u32x4 __attribute__((ext_vector_type(4)));

#define SPLITK 6
#define KSTEP  64                 // K per pipeline stage
#define TILE_ELEMS (32 * KSTEP)   // LDS elements per stage buffer

union FragU { bf16x16 v; u32x4 u[2]; };

// ---- WMMA fragment loaders (per CDNA5 ISA 7.12.2 layouts, wave32) ----

// A: 16x32 bf16. p already points at this lane's row + hi-chunk base.
// VGPR0-3 <- p[0..7] (K koff..koff+7), VGPR4-7 <- p[16..23] (K 16+koff..).
__device__ inline bf16x16 load_frag_a(const bf16* p) {
  FragU f;
  f.u[0] = *(const u32x4*)(p);
  f.u[1] = *(const u32x4*)(p + 16);
  return f.v;
}

// B: 32x16 bf16 from row-major W[N,K] (B = W^T). p already points at this
// lane's column row + koff. VGPR0-3 <- p[0..7], VGPR4-7 <- p[16..23] (same K as A).
__device__ inline bf16x16 load_frag_b(const bf16* p) {
  FragU f;
  f.u[0] = *(const u32x4*)(p);
  f.u[1] = *(const u32x4*)(p + 16);
  return f.v;
}

// Async global->LDS copy of 16 bytes per lane (ASYNCcnt-tracked, no VGPR round trip).
__device__ inline void async_copy16(bf16* lds_dst, const bf16* g_src) {
  unsigned lds = (unsigned)(size_t)lds_dst;   // low 32 bits of generic ptr = LDS offset
  asm volatile("global_load_async_to_lds_b128 %0, %1, off"
               :: "v"(lds), "v"(g_src)
               : "memory");
}

// ---- fused per-step GEMM ----
// blocks 0..47   : gates[64,1536] = Ag @ Wg^T + bias          (K = Kg)
// blocks 48..143 : gkan partials, split-K x6 over Kk, 16 n-tiles of 32 cols
__global__ __launch_bounds__(128) void kanlstm_gemm(
    const bf16* __restrict__ Ag, const bf16* __restrict__ Ak,
    const bf16* __restrict__ Wg, const bf16* __restrict__ Wk,
    const float* __restrict__ biasg,
    float* __restrict__ gates, float* __restrict__ gkan_parts,
    int Kg, int Kk)
{
  __shared__ bf16 sB[4][TILE_ELEMS];   // 4-deep, 32(N) x 64(K) weight tiles (16KB)
  const int tid  = threadIdx.x;
  const int wave = tid >> 5;
  const int lane = tid & 31;

  const bf16* A; const bf16* W; float* C; const float* bias;
  int ldw, ldc, n0, kbeg, kend;
  if ((int)blockIdx.x < 48) {
    A = Ag; W = Wg; C = gates; bias = biasg;
    ldw = Kg; ldc = 1536; n0 = (int)blockIdx.x * 32;
    kbeg = 0; kend = Kg;
  } else {
    int b2  = (int)blockIdx.x - 48;
    int sid = b2 >> 4;                 // split-K slice 0..5
    n0 = (b2 & 15) * 32;
    A = Ak; W = Wk; bias = nullptr;
    C = gkan_parts + (size_t)sid * 64 * 512;
    ldw = Kk; ldc = 512;
    int Ks = Kk / SPLITK;              // 960 or 1536, multiples of 64
    kbeg = sid * Ks; kend = kbeg + Ks;
  }
  const int m0 = wave * 16;

  f32x8 acc0 = {};
  f32x8 acc1 = {};

  // cooperative async-copy addressing: thread -> (row lr, 8-elem chunk lk);
  // each thread copies chunks lk and lk+32 of its row per 64-K stage.
  const int lr = tid >> 2;
  const int lk = (tid & 3) * 8;
  const bf16* gsrc = W + (size_t)(n0 + lr) * ldw + kbeg + lk;
  bf16* lds0 = &sB[0][lr * KSTEP + lk];

  // A-fragment per-lane pointer (A row length == ldw for both GEMMs)
  const int r    = lane & 15;
  const int hi   = (lane >> 4) * 8;
  const bf16* aptr = A + (size_t)(m0 + r) * ldw + kbeg + hi;

  // B-fragment per-lane LDS read offsets
  const int nn   = lane & 15;
  const int koff = (lane >> 4) * 8;
  const bf16* sb0a = &sB[0][nn * KSTEP + koff];          // n-tile 0, ks=0
  const bf16* sb0b = &sB[0][(16 + nn) * KSTEP + koff];   // n-tile 1, ks=0

  const int nIter = (kend - kbeg) >> 6;

  // pipeline prologue: stage 0 -> buffer 0
  async_copy16(lds0, gsrc);
  async_copy16(lds0 + 32, gsrc + 32);

  for (int it = 0; it < nIter - 1; ++it) {
    gsrc += KSTEP;
    bf16* dst = lds0 + ((it + 1) & 3) * TILE_ELEMS;
    async_copy16(dst, gsrc);
    async_copy16(dst + 32, gsrc + 32);
    asm volatile("s_wait_asynccnt 0x2" ::: "memory");   // stage `it` landed (in-order)
    __syncthreads();

    const int bo = (it & 3) * TILE_ELEMS;
    bf16x16 a0  = load_frag_a(aptr);
    bf16x16 b00 = load_frag_b(sb0a + bo);
    bf16x16 b10 = load_frag_b(sb0b + bo);
    acc0 = __builtin_amdgcn_wmma_f32_16x16x32_bf16(
        false, a0, false, b00, (short)0, acc0, false, false);
    acc1 = __builtin_amdgcn_wmma_f32_16x16x32_bf16(
        false, a0, false, b10, (short)0, acc1, false, false);

    bf16x16 a1  = load_frag_a(aptr + 32);
    bf16x16 b01 = load_frag_b(sb0a + bo + 32);
    bf16x16 b11 = load_frag_b(sb0b + bo + 32);
    acc0 = __builtin_amdgcn_wmma_f32_16x16x32_bf16(
        false, a1, false, b01, (short)0, acc0, false, false);
    acc1 = __builtin_amdgcn_wmma_f32_16x16x32_bf16(
        false, a1, false, b11, (short)0, acc1, false, false);

    aptr += KSTEP;
  }

  // pipeline epilogue: last stage
  asm volatile("s_wait_asynccnt 0x0" ::: "memory");
  __syncthreads();
  {
    const int bo = ((nIter - 1) & 3) * TILE_ELEMS;
    bf16x16 a0  = load_frag_a(aptr);
    bf16x16 b00 = load_frag_b(sb0a + bo);
    bf16x16 b10 = load_frag_b(sb0b + bo);
    acc0 = __builtin_amdgcn_wmma_f32_16x16x32_bf16(
        false, a0, false, b00, (short)0, acc0, false, false);
    acc1 = __builtin_amdgcn_wmma_f32_16x16x32_bf16(
        false, a0, false, b10, (short)0, acc1, false, false);

    bf16x16 a1  = load_frag_a(aptr + 32);
    bf16x16 b01 = load_frag_b(sb0a + bo + 32);
    bf16x16 b11 = load_frag_b(sb0b + bo + 32);
    acc0 = __builtin_amdgcn_wmma_f32_16x16x32_bf16(
        false, a1, false, b01, (short)0, acc0, false, false);
    acc1 = __builtin_amdgcn_wmma_f32_16x16x32_bf16(
        false, a1, false, b11, (short)0, acc1, false, false);
  }

  const int rb = (lane >> 4) * 8;
  float bv0 = bias ? bias[n0 + nn]      : 0.0f;
  float bv1 = bias ? bias[n0 + 16 + nn] : 0.0f;
#pragma unroll
  for (int j = 0; j < 8; ++j) {
    C[(size_t)(m0 + rb + j) * ldc + n0 + nn]      = acc0[j] + bv0;
    C[(size_t)(m0 + rb + j) * ldc + n0 + 16 + nn] = acc1[j] + bv1;
  }
}

// ---- per-step activation prep: z=[inp|h] -> bf16 z, silu(z), bsplines(z) ----
__global__ void kanlstm_prep(const float* __restrict__ inp, int inStride,
                             const float* __restrict__ hprev,
                             bf16* __restrict__ Ag, bf16* __restrict__ Ak,
                             int In, int Kz)
{
  int idx = blockIdx.x * blockDim.x + threadIdx.x;
  if (idx >= 64 * Kz) return;
  int b = idx / Kz;
  int k = idx - b * Kz;

  float z = (k < In) ? inp[(size_t)b * inStride + k]
                     : hprev[(size_t)b * 512 + (k - In)];
  Ag[(size_t)b * Kz + k] = (bf16)z;

  float sig = 1.0f / (1.0f + __expf(-z));
  size_t Kk = (size_t)Kz * 9;
  Ak[(size_t)b * Kk + k] = (bf16)(z * sig);   // silu branch

  // Cox-de Boor order-3, knots t[i] = (i-3)*0.4 - 1, i = 0..11
  float bas[11];
#pragma unroll
  for (int i = 0; i < 11; ++i) {
    float t0 = 0.4f * (float)(i - 3) - 1.0f;
    float t1 = 0.4f * (float)(i - 2) - 1.0f;
    bas[i] = (z >= t0 && z < t1) ? 1.0f : 0.0f;
  }
#pragma unroll
  for (int p = 1; p <= 3; ++p) {
    float inv = 1.0f / (0.4f * (float)p);
#pragma unroll
    for (int i = 0; i < 11 - p; ++i) {
      float ti    = 0.4f * (float)(i - 3)     - 1.0f;  // t[i]
      float tip1p = 0.4f * (float)(i + p - 2) - 1.0f;  // t[i+p+1]
      bas[i] = (z - ti) * inv * bas[i] + (tip1p - z) * inv * bas[i + 1];
    }
  }
  bf16* dst = Ak + (size_t)b * Kk + Kz + (size_t)k * 8;
#pragma unroll
  for (int j = 0; j < 8; ++j) dst[j] = (bf16)bas[j];
}

// ---- LSTM cell update (sums the split-K partials of the KAN branch) ----
__global__ void kanlstm_update(const float* __restrict__ gates,
                               const float* __restrict__ gkan_parts,
                               float* __restrict__ h, float* __restrict__ c)
{
  int idx = blockIdx.x * blockDim.x + threadIdx.x;
  if (idx >= 64 * 512) return;
  int b = idx >> 9, j = idx & 511;
  const float* gr = gates + (size_t)b * 1536;
  float ig = 1.0f / (1.0f + __expf(-gr[j]));
  float fg = 1.0f / (1.0f + __expf(-gr[512 + j]));
  float og = 1.0f / (1.0f + __expf(-gr[1024 + j]));
  float g = 0.0f;
#pragma unroll
  for (int s = 0; s < SPLITK; ++s) g += gkan_parts[(size_t)s * 64 * 512 + idx];
  float cn = fg * c[idx] + ig * g;
  c[idx] = cn;
  h[idx] = og * tanhf(cn);
}

// ---- prolog: pack weights to bf16, L2-resident ----
__global__ void kanlstm_pack_wg(const float* __restrict__ w_ih,
                                const float* __restrict__ w_hh,
                                const float* __restrict__ b_ih,
                                const float* __restrict__ b_hh,
                                bf16* __restrict__ Wg, float* __restrict__ biasg,
                                int In, int Kz)
{
  size_t idx = (size_t)blockIdx.x * blockDim.x + threadIdx.x;
  if (idx >= (size_t)1536 * Kz) return;
  int n = (int)(idx / Kz);
  int k = (int)(idx - (size_t)n * Kz);
  float v = (k < In) ? w_ih[(size_t)n * In + k] : w_hh[(size_t)n * 512 + (k - In)];
  Wg[idx] = (bf16)v;
  if (idx < 1536) biasg[idx] = b_ih[idx] + b_hh[idx];
}

__global__ void kanlstm_pack_wk(const float* __restrict__ kb,
                                const float* __restrict__ ks,
                                bf16* __restrict__ Wk, int Kz)
{
  size_t Kk = (size_t)Kz * 9;
  size_t idx = (size_t)blockIdx.x * blockDim.x + threadIdx.x;
  if (idx >= (size_t)512 * Kk) return;
  int n = (int)(idx / Kk);
  size_t k = idx - (size_t)n * Kk;
  float v = (k < (size_t)Kz) ? kb[(size_t)n * Kz + k]
                             : ks[(size_t)n * Kz * 8 + (k - Kz)];
  Wk[idx] = (bf16)v;
}

__global__ void kanlstm_zero(float* __restrict__ p, int n) {
  int idx = blockIdx.x * blockDim.x + threadIdx.x;
  if (idx < n) p[idx] = 0.0f;
}

// ---- epilog: out = h1 @ fc_w^T + fc_b (tiny, fp32) ----
__global__ void kanlstm_fc(const float* __restrict__ h1,
                           const float* __restrict__ fc_w,
                           const float* __restrict__ fc_b,
                           float* __restrict__ out)
{
  int idx = blockIdx.x * blockDim.x + threadIdx.x;
  if (idx >= 64 * 128) return;
  int b = idx >> 7, o = idx & 127;
  float s = fc_b[o];
  const float* hr = h1 + (size_t)b * 512;
  const float* wr = fc_w + (size_t)o * 512;
  for (int j = 0; j < 512; ++j) s += hr[j] * wr[j];
  out[idx] = s;
}

extern "C" void kernel_launch(void* const* d_in, const int* in_sizes, int n_in,
                              void* d_out, int out_size, void* d_ws, size_t ws_size,
                              hipStream_t stream) {
  (void)in_sizes; (void)n_in; (void)out_size; (void)ws_size;

  const float* x          = (const float*)d_in[0];
  const float* w_ih0      = (const float*)d_in[1];
  const float* w_hh0      = (const float*)d_in[2];
  const float* b_ih0      = (const float*)d_in[3];
  const float* b_hh0      = (const float*)d_in[4];
  const float* kan_base0  = (const float*)d_in[5];
  const float* kan_spl0   = (const float*)d_in[6];
  const float* w_ih1      = (const float*)d_in[7];
  const float* w_hh1      = (const float*)d_in[8];
  const float* b_ih1      = (const float*)d_in[9];
  const float* b_hh1      = (const float*)d_in[10];
  const float* kan_base1  = (const float*)d_in[11];
  const float* kan_spl1   = (const float*)d_in[12];
  const float* fc_w       = (const float*)d_in[13];
  const float* fc_b       = (const float*)d_in[14];
  float* out = (float*)d_out;

  const int KZ0 = 640, KZ1 = 1024, KK0 = 5760, KK1 = 9216;

  char* ws = (char*)d_ws;
  size_t off = 0;
  auto alloc = [&](size_t bytes) -> void* {
    void* p = ws + off;
    off = (off + bytes + 255) & ~(size_t)255;
    return p;
  };

  bf16* Wg0    = (bf16*)alloc((size_t)1536 * KZ0 * 2);
  bf16* Wg1    = (bf16*)alloc((size_t)1536 * KZ1 * 2);
  bf16* Wk0    = (bf16*)alloc((size_t)512  * KK0 * 2);
  bf16* Wk1    = (bf16*)alloc((size_t)512  * KK1 * 2);
  float* bg0   = (float*)alloc(1536 * 4);
  float* bg1   = (float*)alloc(1536 * 4);
  bf16* Ag0    = (bf16*)alloc((size_t)64 * KZ0 * 2);
  bf16* Ag1    = (bf16*)alloc((size_t)64 * KZ1 * 2);
  bf16* Ak0    = (bf16*)alloc((size_t)64 * KK0 * 2);
  bf16* Ak1    = (bf16*)alloc((size_t)64 * KK1 * 2);
  float* gates = (float*)alloc((size_t)64 * 1536 * 4);
  float* gkanP = (float*)alloc((size_t)SPLITK * 64 * 512 * 4);
  float* states = (float*)alloc((size_t)4 * 64 * 512 * 4);
  float* h0 = states;
  float* c0 = states + 64 * 512;
  float* h1 = states + 2 * 64 * 512;
  float* c1 = states + 3 * 64 * 512;

  // prolog: pack weights (bf16, stays resident in 192MB L2) + zero state
  kanlstm_pack_wg<<<(1536 * KZ0) / 256, 256, 0, stream>>>(w_ih0, w_hh0, b_ih0, b_hh0, Wg0, bg0, 128, KZ0);
  kanlstm_pack_wg<<<(1536 * KZ1) / 256, 256, 0, stream>>>(w_ih1, w_hh1, b_ih1, b_hh1, Wg1, bg1, 512, KZ1);
  kanlstm_pack_wk<<<(512 * KK0) / 256, 256, 0, stream>>>(kan_base0, kan_spl0, Wk0, KZ0);
  kanlstm_pack_wk<<<(512 * KK1) / 256, 256, 0, stream>>>(kan_base1, kan_spl1, Wk1, KZ1);
  kanlstm_zero<<<(4 * 64 * 512) / 256, 256, 0, stream>>>(states, 4 * 64 * 512);

  const int GEMM_BLOCKS = 48 + 16 * SPLITK;   // 144

  // recurrence: stream order provides cross-step dependencies
  for (int t = 0; t < 256; ++t) {
    // layer 0: input x[:, t, :], row stride 256*128
    kanlstm_prep<<<(64 * KZ0) / 256, 256, 0, stream>>>(x + (size_t)t * 128, 256 * 128, h0, Ag0, Ak0, 128, KZ0);
    kanlstm_gemm<<<GEMM_BLOCKS, 128, 0, stream>>>(Ag0, Ak0, Wg0, Wk0, bg0, gates, gkanP, KZ0, KK0);
    kanlstm_update<<<(64 * 512) / 256, 256, 0, stream>>>(gates, gkanP, h0, c0);
    // layer 1: input h0
    kanlstm_prep<<<(64 * KZ1) / 256, 256, 0, stream>>>(h0, 512, h1, Ag1, Ak1, 512, KZ1);
    kanlstm_gemm<<<GEMM_BLOCKS, 128, 0, stream>>>(Ag1, Ak1, Wg1, Wk1, bg1, gates, gkanP, KZ1, KK1);
    kanlstm_update<<<(64 * 512) / 256, 256, 0, stream>>>(gates, gkanP, h1, c1);
  }

  kanlstm_fc<<<(64 * 128) / 256, 256, 0, stream>>>(h1, fc_w, fc_b, out);
}
